// NodeGNNModel_89704686944683
// MI455X (gfx1250) — matrix operand order repaired and checked by reading, compiled, not run on hardware
//
#include <hip/hip_runtime.h>

#define N_NODES 50000
#define N_EDGES 600000
#define IN_DIM  256
#define HID     128
#define OUT_DIM 10
#define N_LAYERS 3
#define BN_EPS  1e-5f

typedef __attribute__((ext_vector_type(16))) __bf16 v16bf;
typedef __attribute__((ext_vector_type(8)))  float  v8f;

union BF16x16 { v16bf v; uint4 q[2]; };

__device__ __forceinline__ unsigned short f2bf(float f) {
    unsigned u = __float_as_uint(f);
    unsigned r = u + 0x7FFFu + ((u >> 16) & 1u);   // round-to-nearest-even
    return (unsigned short)(r >> 16);
}

// ---------------------------------------------------------------- utilities
__global__ void zero_f32_kernel(float* p, int n) {
    int i = blockIdx.x * blockDim.x + threadIdx.x;
    int stride = gridDim.x * blockDim.x;
    for (; i < n; i += stride) p[i] = 0.0f;
}

__global__ void cvt_bf16_kernel(const float* __restrict__ x,
                                unsigned short* __restrict__ xb, int n) {
    int i = blockIdx.x * blockDim.x + threadIdx.x;
    if (i < n) xb[i] = f2bf(x[i]);
}

// Pre-pack a [K x 128] f32 weight matrix into CDNA5 WMMA B-fragment order (bf16 pairs).
// Element index = ((kt*8 + ntile)*32 + lane)*8 + j ; pair = W[k, c], W[k+1, c]
// with k = kt*32 + 8*(lane>>4) + 2*j + (j>=4 ? 8 : 0), c = ntile*16 + (lane&15).
__global__ void prepack_b_kernel(const float* __restrict__ W,
                                 unsigned* __restrict__ out, int ktiles) {
    int id = blockIdx.x * blockDim.x + threadIdx.x;
    int total = ktiles * 8 * 32 * 8;
    if (id >= total) return;
    int j  = id & 7;
    int l  = (id >> 3) & 31;
    int nt = (id >> 8) & 7;
    int kt = id >> 11;
    int k = kt * 32 + ((l >> 4) * 8) + 2 * j + ((j >> 2) * 8);
    int c = nt * 16 + (l & 15);
    unsigned lo = f2bf(W[k * HID + c]);
    unsigned hi = f2bf(W[(k + 1) * HID + c]);
    out[id] = lo | (hi << 16);
}

// ---------------------------------------------------------------- WMMA GEMM
// C[M x 128] = A[M x K] * B ; A bf16 row-major, B pre-packed fragments.
// One wave = one 16x16 tile; block = 256 threads = 8 waves = 16 rows x 128 cols.
template <int K, bool BIAS, bool RELU, bool WRITE_BF>
__global__ void gemm_bf16_kernel(const unsigned short* __restrict__ A,
                                 const unsigned* __restrict__ Bpack,
                                 const float* __restrict__ bias,
                                 float* __restrict__ Cf,
                                 unsigned short* __restrict__ Cb) {
    const int wave = threadIdx.x >> 5;     // column tile 0..7
    const int lane = threadIdx.x & 31;
    const int r0   = blockIdx.x * 16;
    const int arow = r0 + (lane & 15);     // A-fragment row for this lane
    const int ksel = lane >> 4;

    v8f acc = {};
    constexpr int KT = K >> 5;
#pragma unroll
    for (int kt = 0; kt < KT; ++kt) {
        BF16x16 a, b;
        const uint4* ap = (const uint4*)(A + (size_t)arow * K + kt * 32 + ksel * 8);
        a.q[0] = ap[0];          // K chunk [8*ksel, 8*ksel+8)
        a.q[1] = ap[2];          // K chunk [16+8*ksel, 16+8*ksel+8)
        const uint4* bp = (const uint4*)(Bpack + (((size_t)kt * 8 + wave) * 32 + lane) * 8);
        b.q[0] = bp[0];
        b.q[1] = bp[1];
        acc = __builtin_amdgcn_wmma_f32_16x16x32_bf16(false, a.v, false, b.v,
                                                      (short)0, acc, false, false);
    }

    const int col = wave * 16 + (lane & 15);
#pragma unroll
    for (int j = 0; j < 8; ++j) {
        int r = r0 + j + 8 * ksel;         // C layout: VGPR j -> M = j (+8 for hi lanes)
        float v = acc[j];
        if (BIAS) v += bias[col];
        if (RELU) v = fmaxf(v, 0.0f);
        Cf[(size_t)r * HID + col] = v;
        if (WRITE_BF) Cb[(size_t)r * HID + col] = f2bf(v);
    }
}

// ---------------------------------------------------------------- graph ops
__global__ void deg_kernel(const long long* __restrict__ dst, float* __restrict__ deg) {
    int e = blockIdx.x * blockDim.x + threadIdx.x;
    if (e < N_EDGES) atomicAdd(&deg[(int)dst[e]], 1.0f);
}

__global__ void dinv_kernel(const float* __restrict__ deg, float* __restrict__ dinv) {
    int n = blockIdx.x * blockDim.x + threadIdx.x;
    if (n < N_NODES) dinv[n] = rsqrtf(deg[n] + 1.0f);
}

// One wave per edge, lane handles 4 channels. Gather t[src]*norm, atomic-add to agg[dst].
__global__ void scatter_kernel(const long long* __restrict__ src,
                               const long long* __restrict__ dst,
                               const float* __restrict__ dinv,
                               const float* __restrict__ t,
                               float* __restrict__ agg) {
    long long gid = (long long)blockIdx.x * blockDim.x + threadIdx.x;
    int e = (int)(gid >> 5);
    if (e >= N_EDGES) return;
    int lane = (int)gid & 31;
    int s = (int)src[e];
    int d = (int)dst[e];
    float sc = dinv[s] * dinv[d];
    const float4 v = *(const float4*)(t + (size_t)s * HID + lane * 4);
    float* a = agg + (size_t)d * HID + lane * 4;
    atomicAdd(a + 0, v.x * sc);
    atomicAdd(a + 1, v.y * sc);
    atomicAdd(a + 2, v.z * sc);
    atomicAdd(a + 3, v.w * sc);
}

// agg += t * dinv^2 + bc ; simultaneously accumulate per-channel sum & sumsq for BN.
// Launch with gridDim*blockDim = 64000 (multiple of 128 so each thread owns one channel).
__global__ void selfloop_stats_kernel(const float* __restrict__ t,
                                      const float* __restrict__ dinv,
                                      const float* __restrict__ bc,
                                      float* __restrict__ agg,
                                      float* __restrict__ sums) {
    const int total = N_NODES * HID;
    int base = blockIdx.x * blockDim.x + threadIdx.x;
    int stride = gridDim.x * blockDim.x;
    int c = base & (HID - 1);
    float b = bc[c];
    float s = 0.0f, s2 = 0.0f;
    for (int i = base; i < total; i += stride) {
        int n = i >> 7;
        float di = dinv[n];
        float v = agg[i] + t[i] * (di * di) + b;
        agg[i] = v;
        s += v;
        s2 += v * v;
    }
    atomicAdd(&sums[c], s);
    atomicAdd(&sums[HID + c], s2);
}

// h = relu((agg - mu) * rsqrt(var+eps) * gamma + beta) [+ h_res] ; also bf16 copy.
__global__ void bn_apply_kernel(const float* __restrict__ agg,
                                const float* __restrict__ sums,
                                const float* __restrict__ gamma,
                                const float* __restrict__ beta,
                                float* __restrict__ h,
                                unsigned short* __restrict__ hb,
                                int addRes) {
    const int total = N_NODES * HID;
    const float invN = 1.0f / (float)N_NODES;
    int i = blockIdx.x * blockDim.x + threadIdx.x;
    int stride = gridDim.x * blockDim.x;
    for (; i < total; i += stride) {
        int c = i & (HID - 1);
        float mu  = sums[c] * invN;
        float var = sums[HID + c] * invN - mu * mu;
        float v = (agg[i] - mu) * rsqrtf(var + BN_EPS) * gamma[c] + beta[c];
        v = fmaxf(v, 0.0f);
        if (addRes) v += h[i];
        h[i] = v;
        hb[i] = f2bf(v);
    }
}

// out[N,10] = h[N,128] @ W_out[128,10] + b_out  (tiny; L1/L2-cached scalar dot)
__global__ void out_kernel(const float* __restrict__ h,
                           const float* __restrict__ Wout,
                           const float* __restrict__ bout,
                           float* __restrict__ out) {
    int gid = blockIdx.x * blockDim.x + threadIdx.x;
    if (gid >= N_NODES * OUT_DIM) return;
    int n = gid / OUT_DIM;
    int c = gid - n * OUT_DIM;
    float acc = bout[c];
    const float* hr = h + (size_t)n * HID;
#pragma unroll 8
    for (int k = 0; k < HID; ++k) acc += hr[k] * Wout[k * OUT_DIM + c];
    out[gid] = acc;
}

// ---------------------------------------------------------------- launch
extern "C" void kernel_launch(void* const* d_in, const int* in_sizes, int n_in,
                              void* d_out, int out_size, void* d_ws, size_t ws_size,
                              hipStream_t stream) {
    const float*     x     = (const float*)d_in[0];
    const long long* ei    = (const long long*)d_in[1];   // int64 [2, E]
    const float*     W_in  = (const float*)d_in[2];
    const float*     b_in  = (const float*)d_in[3];
    const float*     Wc    = (const float*)d_in[4];
    const float*     bc    = (const float*)d_in[5];
    const float*     gamma = (const float*)d_in[6];
    const float*     beta  = (const float*)d_in[7];
    const float*     W_out = (const float*)d_in[8];
    const float*     b_out = (const float*)d_in[9];
    float* out = (float*)d_out;

    const long long* src = ei;
    const long long* dst = ei + N_EDGES;

    // Workspace carve-out (256B aligned slabs).
    char* ws = (char*)d_ws;
    size_t off = 0;
    auto carve = [&](size_t bytes) -> char* {
        char* p = ws + off;
        off = (off + bytes + 255) & ~(size_t)255;
        return p;
    };
    unsigned short* xb   = (unsigned short*)carve(2ull * N_NODES * IN_DIM);
    float*          h    = (float*)carve(4ull * N_NODES * HID);
    unsigned short* hb   = (unsigned short*)carve(2ull * N_NODES * HID);
    float*          t    = (float*)carve(4ull * N_NODES * HID);
    float*          agg  = (float*)carve(4ull * N_NODES * HID + 4 * 2 * HID); // + sums tail
    float*          sums = agg + (size_t)N_NODES * HID;                        // [128 sum | 128 sumsq]
    float*          deg  = (float*)carve(4ull * N_NODES);
    float*          dinv = (float*)carve(4ull * N_NODES);
    unsigned*       BpIn = (unsigned*)carve(4ull * (IN_DIM / 32) * 2048);      // 16384 u32
    unsigned*       BpC  = (unsigned*)carve(4ull * N_LAYERS * (HID / 32) * 2048); // 3*8192 u32

    // --- one-time prep ---
    zero_f32_kernel<<<256, 256, 0, stream>>>(deg, N_NODES);
    cvt_bf16_kernel<<<(N_NODES * IN_DIM + 255) / 256, 256, 0, stream>>>(x, xb, N_NODES * IN_DIM);
    prepack_b_kernel<<<(16384 + 255) / 256, 256, 0, stream>>>(W_in, BpIn, IN_DIM / 32);
    for (int i = 0; i < N_LAYERS; ++i)
        prepack_b_kernel<<<(8192 + 255) / 256, 256, 0, stream>>>(
            Wc + (size_t)i * HID * HID, BpC + (size_t)i * 8192, HID / 32);
    deg_kernel<<<(N_EDGES + 255) / 256, 256, 0, stream>>>(dst, deg);
    dinv_kernel<<<(N_NODES + 255) / 256, 256, 0, stream>>>(deg, dinv);

    // --- input layer: h = relu(x @ W_in + b_in) ---
    gemm_bf16_kernel<IN_DIM, true, true, true>
        <<<N_NODES / 16, 256, 0, stream>>>(xb, BpIn, b_in, h, hb);

    // --- GCN layers ---
    for (int i = 0; i < N_LAYERS; ++i) {
        zero_f32_kernel<<<4096, 256, 0, stream>>>(agg, N_NODES * HID + 2 * HID);
        gemm_bf16_kernel<HID, false, false, false>
            <<<N_NODES / 16, 256, 0, stream>>>(hb, BpC + (size_t)i * 8192, nullptr, t, nullptr);
        scatter_kernel<<<(N_EDGES * 32) / 256, 256, 0, stream>>>(src, dst, dinv, t, agg);
        selfloop_stats_kernel<<<250, 256, 0, stream>>>(t, dinv, bc + (size_t)i * HID, agg, sums);
        bn_apply_kernel<<<1000, 256, 0, stream>>>(agg, sums, gamma + (size_t)i * HID,
                                                  beta + (size_t)i * HID, h, hb, i > 0);
    }

    // --- output projection ---
    out_kernel<<<(N_NODES * OUT_DIM + 255) / 256, 256, 0, stream>>>(h, W_out, b_out, out);
}
